// RNN_11424613008107
// MI455X (gfx1250) — matrix operand order repaired and checked by reading, not compiled, not run on hardware
//
#include <hip/hip_runtime.h>
#include <math.h>

// Problem dims (from reference)
#define V_   32000
#define E_   512
#define H_   512
#define MID_ 128
#define B_   16
#define S_   512
#define EH_  1024   // E_ + H_

typedef __attribute__((ext_vector_type(16))) _Float16 v16h;
typedef __attribute__((ext_vector_type(8)))  _Float16 v8h;
typedef __attribute__((ext_vector_type(8)))  float    v8f;
typedef __attribute__((ext_vector_type(4)))  int      v4i;

union V16U { v16h v; v8h h[2]; };

// CDNA5 async global->LDS copy path (guarded: falls back to sync copy)
#if __has_builtin(__builtin_amdgcn_global_load_async_to_lds_b128) && \
    __has_builtin(__builtin_amdgcn_s_wait_asynccnt)
#define USE_ASYNC_LDS 1
#else
#define USE_ASYNC_LDS 0
#endif

#define GLOBAL_AS __attribute__((address_space(1)))
#define LDS_AS    __attribute__((address_space(3)))

// ---------------------------------------------------------------------------
// WMMA fragment loaders (layouts per CDNA5 ISA §7.12.2, wave32)
// ---------------------------------------------------------------------------

// A-matrix 16x32 f16 fragment from row-major [row][k] tile (ld = row stride).
__device__ __forceinline__ v16h load_a_frag(const _Float16* tile, int ld, int lane) {
  const int row  = lane & 15;
  const int half = lane >> 4;
  const _Float16* p = tile + row * ld + half * 8;
  V16U u;
  u.h[0] = *(const v8h*)(p);        // K = half*8 + 0..7
  u.h[1] = *(const v8h*)(p + 16);   // K = half*8 + 16..23
  return u.v;
}

// B-matrix 32x16 f16 fragment from K-major storage Bt[col][k] (ld = K stride).
__device__ __forceinline__ v16h load_b_frag(const _Float16* bt, int ld, int col,
                                            int k0, int lane) {
  const int half = lane >> 4;
  const _Float16* p = bt + col * ld + k0 + half * 8;
  V16U u;
  u.h[0] = *(const v8h*)(p);
  u.h[1] = *(const v8h*)(p + 16);
  return u.v;
}

// async (or fallback) 16-byte global -> LDS copy of 8 f16
__device__ __forceinline__ void copy8h_to_lds(_Float16* lds_dst, const _Float16* gsrc) {
#if USE_ASYNC_LDS
  __builtin_amdgcn_global_load_async_to_lds_b128(
      (GLOBAL_AS v4i*)(gsrc),
      (LDS_AS v4i*)(lds_dst), 0, 0);
#else
  *(v8h*)lds_dst = *(const v8h*)gsrc;
#endif
}

__device__ __forceinline__ void wait_async_copies() {
#if USE_ASYNC_LDS
  __builtin_amdgcn_s_wait_asynccnt(0);
#endif
}

// ---------------------------------------------------------------------------
// Prep kernels: fp32 -> f16 conversion / transpose / embedding gather
// ---------------------------------------------------------------------------

__global__ __launch_bounds__(256)
void convert_embed_kernel(const float* __restrict__ src, _Float16* __restrict__ dst, int n) {
  int i = (blockIdx.x * 256 + threadIdx.x) * 8;
  if (i + 8 <= n) {
    #pragma unroll
    for (int j = 0; j < 8; ++j) dst[i + j] = (_Float16)src[i + j];
  }
}

// src: (K, N) f32 row-major  ->  dst: (N, K) f16 row-major (K-major for WMMA B)
__global__ __launch_bounds__(256)
void transpose_f16_kernel(const float* __restrict__ src, _Float16* __restrict__ dst,
                          int K, int N) {
  int i = blockIdx.x * 256 + threadIdx.x;
  if (i < N * K) {
    int n = i / K, k = i % K;
    dst[i] = (_Float16)src[k * N + n];
  }
}

// xf[(b*S+t)*E + e] = embf[ids[b*S+t]*E + e]
__global__ __launch_bounds__(256)
void gather_kernel(const int* __restrict__ ids, const _Float16* __restrict__ embf,
                   _Float16* __restrict__ xf) {
  int i = (blockIdx.x * 256 + threadIdx.x) * 8;   // over B*S*E, exact grid
  int row = i >> 9;        // E_ == 512
  int e   = i & 511;
  int id  = ids[row];
  *(v8h*)(xf + i) = *(const v8h*)(embf + (size_t)id * E_ + e);
}

// ---------------------------------------------------------------------------
// Recurrent scan: single workgroup (8 waves), all 16 batch rows = one M-tile.
// Double-buffered async x staging: copy for step t+1 overlaps step t compute.
//   z     = [x_t | h] (16x1024) @ W1 (1024x128) + b1      -> 8 waves x 1 tile
//   h_new = gelu(z)   (16x128)  @ W2 (128x512)            -> 8 waves x 4 tiles
// ---------------------------------------------------------------------------

__device__ __forceinline__ void stage_x_async(_Float16* dstX, const _Float16* xf,
                                              int t, int tid) {
  #pragma unroll
  for (int c = 0; c < 4; ++c) {
    int i = (tid + c * 256) * 8;            // over B_*E_ = 8192 halves
    int b = i >> 9, e = i & 511;
    copy8h_to_lds(dstX + b * E_ + e, xf + ((size_t)(b * S_ + t)) * E_ + e);
  }
}

__global__ __launch_bounds__(256)
void scan_kernel(const _Float16* __restrict__ xf,      // (B*S, E) f16
                 const float*    __restrict__ hidden,  // (B, H) f32
                 const _Float16* __restrict__ w1t,     // (MID, EH) K-major f16
                 const float*    __restrict__ b1,      // (MID)
                 const _Float16* __restrict__ w2t,     // (E, MID) K-major f16
                 _Float16* __restrict__ outf)          // (B*S, E) f16
{
  __shared__ _Float16 Xbuf[2][B_ * E_];  // double-buffered x_t (2 x 16 KB)
  __shared__ _Float16 Hb[B_ * H_];       // hidden state (16 KB)
  __shared__ _Float16 A2[B_ * MID_];     // gelu(z) (4 KB)

  const int tid  = threadIdx.x;
  const int lane = tid & 31;
  const int w    = tid >> 5;    // 8 waves
  const int half = lane >> 4;
  const int ln   = lane & 15;

  // h0 from `hidden`
  for (int i = tid; i < B_ * H_; i += 256) {
    int b = i >> 9, j = i & 511;
    Hb[b * H_ + j] = (_Float16)hidden[b * H_ + j];
  }

  // preload x_0
  stage_x_async(Xbuf[0], xf, 0, tid);
  wait_async_copies();
  __syncthreads();

  for (int t = 0; t < S_; ++t) {
    const _Float16* Xc = Xbuf[t & 1];

    // ---- kick off async staging of x_{t+1} into the other buffer
    if (t + 1 < S_) stage_x_async(Xbuf[(t + 1) & 1], xf, t + 1, tid);

    // ---- phase 1: z tile per wave (MID = 8 tiles of 16)
    {
      const int col = w * 16 + ln;
      const float bias = b1[col];
      v8f c;
      #pragma unroll
      for (int i = 0; i < 8; ++i) c[i] = bias;
      #pragma unroll 4
      for (int kk = 0; kk < E_ / 32; ++kk) {           // K = 0..511 from x_t
        v16h a = load_a_frag(Xc + kk * 32, E_, lane);
        v16h b = load_b_frag(w1t, EH_, col, kk * 32, lane);
        c = __builtin_amdgcn_wmma_f32_16x16x32_f16(false, a, false, b,
                                                   (short)0, c, false, false);
      }
      #pragma unroll 4
      for (int kk = 0; kk < H_ / 32; ++kk) {           // K = 512..1023 from h
        v16h a = load_a_frag(Hb + kk * 32, H_, lane);
        v16h b = load_b_frag(w1t, EH_, col, E_ + kk * 32, lane);
        c = __builtin_amdgcn_wmma_f32_16x16x32_f16(false, a, false, b,
                                                   (short)0, c, false, false);
      }
      // exact GELU: 0.5*z*(1+erf(z/sqrt(2))), then f16 into A2
      #pragma unroll
      for (int i = 0; i < 8; ++i) {
        float z = c[i];
        float g = 0.5f * z * (1.0f + erff(z * 0.70710678118654752f));
        A2[(i + half * 8) * MID_ + col] = (_Float16)g;
      }
    }
    __syncthreads();

    // ---- phase 2: h_new; wave covers 4 column tiles (64 of 512 cols)
    {
      v8f acc[4];
      #pragma unroll
      for (int q = 0; q < 4; ++q)
        #pragma unroll
        for (int i = 0; i < 8; ++i) acc[q][i] = 0.0f;

      #pragma unroll
      for (int kk = 0; kk < MID_ / 32; ++kk) {
        v16h a = load_a_frag(A2 + kk * 32, MID_, lane);
        #pragma unroll
        for (int q = 0; q < 4; ++q) {
          int col = (w * 4 + q) * 16 + ln;
          v16h b = load_b_frag(w2t, MID_, col, kk * 32, lane);
          acc[q] = __builtin_amdgcn_wmma_f32_16x16x32_f16(false, a, false, b,
                                                          (short)0, acc[q], false, false);
        }
      }
      #pragma unroll
      for (int q = 0; q < 4; ++q) {
        int col = (w * 4 + q) * 16 + ln;
        #pragma unroll
        for (int i = 0; i < 8; ++i) {
          int m = i + half * 8;                       // batch row
          _Float16 hv = (_Float16)acc[q][i];
          Hb[m * H_ + col] = hv;                      // h for next step
          outf[((size_t)(m * S_ + t)) * E_ + col] = hv;
        }
      }
    }
    // make x_{t+1} staging visible to everyone before the next iteration
    wait_async_copies();
    __syncthreads();
  }
}

// ---------------------------------------------------------------------------
// Classifier: logits = out @ embed.T + cls_b   (8192x512 @ 512x32000)
// Block = 256 thr = 8 waves; block tile M=64 x N=256.
// A panel (64x512 f16 = 64 KB) async-staged to LDS; wave = 2 M-tiles x 4 N-tiles.
// embed row-major (V,E) is already K-major for B = embed.T.
// Logits stored non-temporally so the 1.05 GB stream doesn't evict embed in L2.
// ---------------------------------------------------------------------------

__global__ __launch_bounds__(256)
void classifier_kernel(const _Float16* __restrict__ outf,  // (8192, 512)
                       const _Float16* __restrict__ embf,  // (V, 512)
                       const float*    __restrict__ cls_b, // (V)
                       float* __restrict__ logits) {       // (8192, V)
  __shared__ _Float16 Ash[64 * E_];   // 64 KB

  const int tid  = threadIdx.x;
  const int lane = tid & 31;
  const int w    = tid >> 5;
  const int half = lane >> 4;
  const int ln   = lane & 15;
  const int mblk = blockIdx.y * 64;
  const int n0   = blockIdx.x * 256 + (w >> 1) * 64;
  const int mt0  = (w & 1) * 2;       // wave's first M tile (of 4 per block)

  // prefetch the embedding rows this lane will consume (global_prefetch_b8)
  #pragma unroll
  for (int q = 0; q < 4; ++q)
    __builtin_prefetch(embf + (size_t)(n0 + q * 16 + ln) * E_, 0, 3);

  // async-stage A panel: 64 rows x 512 K, 4096 x 16B chunks, 16 per thread
  {
    const _Float16* gsrc = outf + (size_t)mblk * E_;
    #pragma unroll
    for (int c = 0; c < 16; ++c) {
      int off = (tid + c * 256) * 8;
      copy8h_to_lds(Ash + off, gsrc + off);
    }
    wait_async_copies();
  }
  __syncthreads();

  v8f acc[8];                         // [mi][q]
  #pragma unroll
  for (int q = 0; q < 4; ++q) {
    float bias = cls_b[n0 + q * 16 + ln];
    #pragma unroll
    for (int i = 0; i < 8; ++i) { acc[q][i] = bias; acc[4 + q][i] = bias; }
  }

  #pragma unroll 2
  for (int kk = 0; kk < E_ / 32; ++kk) {
    v16h a0 = load_a_frag(Ash + (mt0 + 0) * 16 * E_ + kk * 32, E_, lane);
    v16h a1 = load_a_frag(Ash + (mt0 + 1) * 16 * E_ + kk * 32, E_, lane);
    #pragma unroll
    for (int q = 0; q < 4; ++q) {
      int col = n0 + q * 16 + ln;
      v16h b = load_b_frag(embf, E_, col, kk * 32, lane);
      acc[q]     = __builtin_amdgcn_wmma_f32_16x16x32_f16(false, a0, false, b,
                                                          (short)0, acc[q], false, false);
      acc[4 + q] = __builtin_amdgcn_wmma_f32_16x16x32_f16(false, a1, false, b,
                                                          (short)0, acc[4 + q], false, false);
    }
  }

  #pragma unroll
  for (int mi = 0; mi < 2; ++mi) {
    #pragma unroll
    for (int q = 0; q < 4; ++q) {
      int col = n0 + q * 16 + ln;
      #pragma unroll
      for (int i = 0; i < 8; ++i) {
        int r = mblk + (mt0 + mi) * 16 + i + half * 8;
        __builtin_nontemporal_store(acc[mi * 4 + q][i], &logits[(size_t)r * V_ + col]);
      }
    }
  }
}

// ---------------------------------------------------------------------------

extern "C" void kernel_launch(void* const* d_in, const int* in_sizes, int n_in,
                              void* d_out, int out_size, void* d_ws, size_t ws_size,
                              hipStream_t stream) {
  const int*   ids    = (const int*)  d_in[0];   // (B, S)
  const float* hidden = (const float*)d_in[1];   // (B, 1, H)
  const float* embed  = (const float*)d_in[2];   // (V, E)
  const float* W1     = (const float*)d_in[3];   // (EH, MID)
  const float* b1     = (const float*)d_in[4];   // (MID)
  const float* W2     = (const float*)d_in[5];   // (MID, E)
  const float* cls_b  = (const float*)d_in[6];   // (V)
  float* logits = (float*)d_out;                 // (B, S, V) f32

  // Workspace layout (all offsets 16B-aligned)
  char* ws = (char*)d_ws;
  size_t off = 0;
  _Float16* embf = (_Float16*)(ws + off); off += (size_t)V_ * E_ * 2;      // 31.25 MB
  _Float16* w1t  = (_Float16*)(ws + off); off += (size_t)MID_ * EH_ * 2;   // 256 KB
  _Float16* w2t  = (_Float16*)(ws + off); off += (size_t)E_ * MID_ * 2;    // 128 KB
  _Float16* xf   = (_Float16*)(ws + off); off += (size_t)B_ * S_ * E_ * 2; // 8 MB
  _Float16* outf = (_Float16*)(ws + off); off += (size_t)B_ * S_ * E_ * 2; // 8 MB

  // 1) f16 conversions + weight transposes (K-major for WMMA B fragments)
  convert_embed_kernel<<<(V_ * E_) / (256 * 8), 256, 0, stream>>>(embed, embf, V_ * E_);
  transpose_f16_kernel<<<(MID_ * EH_ + 255) / 256, 256, 0, stream>>>(W1, w1t, EH_, MID_);
  transpose_f16_kernel<<<(E_ * MID_ + 255) / 256, 256, 0, stream>>>(W2, w2t, MID_, E_);

  // 2) embedding gather (f16)
  gather_kernel<<<(B_ * S_ * E_) / (256 * 8), 256, 0, stream>>>(ids, embf, xf);

  // 3) sequential recurrence: one workgroup, WMMA tiles, double-buffered async x
  scan_kernel<<<1, 256, 0, stream>>>(xf, hidden, w1t, b1, w2t, outf);

  // 4) big classifier GEMM: 268 GFLOP f16-WMMA, writes 1.05 GB fp32 (NT stores)
  dim3 cgrid(V_ / 256, (B_ * S_) / 64);
  classifier_kernel<<<cgrid, 256, 0, stream>>>(outf, embf, cls_b, logits);
}
